// BiLSTM_CRF_Model_57509612093601
// MI455X (gfx1250) — matrix operand order repaired and checked by reading, not compiled, not run on hardware
//
#include <hip/hip_runtime.h>
#include <hip/hip_bf16.h>

typedef _Float16 f16;
typedef __attribute__((ext_vector_type(16))) _Float16 v16h;
typedef __attribute__((ext_vector_type(8)))  _Float16 v8h;
typedef __attribute__((ext_vector_type(8)))  float    v8f;
typedef __attribute__((ext_vector_type(4)))  unsigned int u32x4;
typedef __attribute__((ext_vector_type(8)))  unsigned int u32x8;

#define T_LEN 512
#define BATCH 128
#define EMB   64
#define HID   128
#define NTAG  32
#define KDIM  192               // EMB + HID (fused GEMM K)
#define XH2   384               // 2*64 x double-buf + 2*128 h double-buf (f16)
#define NEGV  -10000.0f

// ---------------------------------------------------------------------------
// Kernel 1a: repack [Wih|Whh] into f16, gate-interleaved column permutation.
// Permuted col p: j = p>>6 (hid group of 16), gate = (p>>4)&3, u = p&15.
// orig z-col = gate*128 + j*16 + u.   Wp[dir][p][k], k<64 -> Wih, else Whh.
// ---------------------------------------------------------------------------
__global__ __launch_bounds__(192) void prep_kernel(
    const float* __restrict__ Wih_f, const float* __restrict__ Whh_f, const float* __restrict__ b_f,
    const float* __restrict__ Wih_b, const float* __restrict__ Whh_b, const float* __restrict__ b_b,
    f16* __restrict__ Wp, float* __restrict__ bp)
{
  int dir = blockIdx.x >> 9;
  int p   = blockIdx.x & 511;
  int k   = threadIdx.x;
  const float* Wih = dir ? Wih_b : Wih_f;
  const float* Whh = dir ? Whh_b : Whh_f;
  const float* bb  = dir ? b_b   : b_f;
  int gate = (p >> 4) & 3, j = p >> 6, u = p & 15;
  int orig = gate * 128 + j * 16 + u;
  float w = (k < 64) ? Wih[orig * 64 + k] : Whh[orig * 128 + (k - 64)];
  Wp[((size_t)dir * 512 + p) * KDIM + k] = (f16)w;
  if (k == 0) bp[dir * 512 + p] = bb[orig];
}

// Kernel 1b: embedding table f32 -> f16 (enables raw async byte-copy per step)
__global__ __launch_bounds__(256) void emb_prep_kernel(
    const float* __restrict__ emb, f16* __restrict__ embh)
{
  size_t i = (size_t)blockIdx.x * 256 + threadIdx.x;
  embh[i] = (f16)emb[i];
}

// ---------------------------------------------------------------------------
// Kernel 2: persistent biLSTM. grid = 2 (dir), block = 512 (16 waves).
// LDS: weights f16 [512][192] (192KB, filled by TDM tensor_load_to_lds)
//    + xh f16 [128][384] (96KB: double-buffered x and h) = 288KB.
// c-state in VGPRs. Next step's embedding rows prefetched with
// GLOBAL_LOAD_ASYNC_TO_LDS_B128 (ASYNCcnt) while WMMAs of current step run.
// Accumulators start from WMMA's inline-zero SRC2; bias folded into epilogue.
// ---------------------------------------------------------------------------
__global__ __launch_bounds__(512) void lstm_kernel(
    const int* __restrict__ sent, const f16* __restrict__ embh,
    const f16* __restrict__ Wp, const float* __restrict__ bp,
    f16* __restrict__ h_out)
{
  extern __shared__ char smem[];
  f16* Wl = (f16*)smem;                          // 512*192 f16 = 196608 B
  f16* xh = (f16*)(smem + 512 * KDIM * 2);       // 128*384 f16 = 98304 B
  const int dir = blockIdx.x;
  const f16*   Wg = Wp + (size_t)dir * 512 * KDIM;
  const float* bg = bp + dir * 512;
  f16* hg = h_out + (size_t)dir * T_LEN * BATCH * HID;

  const int tid  = threadIdx.x;
  const int wave = tid >> 5, lane = tid & 31;
  const int half = lane >> 4, lq = lane & 15;
  const int row4 = tid >> 2, seg = tid & 3;      // embedding copy mapping

  // --- one-shot weight fill via Tensor Data Mover (wave 0 issues) ---------
  if (wave == 0) {
    unsigned long long ga = (unsigned long long)(uintptr_t)Wg;
    unsigned ldsa = (unsigned)(uintptr_t)Wl;
    const unsigned ndw = 512u * KDIM * 2u / 4u;  // 49152 dwords
    u32x4 g0;
    g0[0] = 1u;                                          // count=1, user D#
    g0[1] = ldsa;                                        // lds_addr
    g0[2] = (unsigned)(ga & 0xFFFFFFFFu);                // global_addr[31:0]
    g0[3] = (unsigned)((ga >> 32) & 0x1FFFFFFu) | (2u << 30); // addr[56:32], type=2
    u32x8 g1;
    g1[0] = (2u << 16);                                  // data_size = 4B
    g1[1] = (ndw & 0xFFFFu) << 16;                       // tensor_dim0[15:0]
    g1[2] = ((ndw >> 16) & 0xFFFFu) | (1u << 16);        // tensor_dim0 hi, tensor_dim1=1
    g1[3] = (ndw & 0xFFFFu) << 16;                       // tile_dim0
    g1[4] = 1u;                                          // tile_dim1 = 1
    g1[5] = ndw;                                         // tensor_dim0_stride lo
    g1[6] = 0u;
    g1[7] = 0u;
    asm volatile("tensor_load_to_lds %0, %1" :: "s"(g0), "s"(g1) : "memory");
    __builtin_amdgcn_s_wait_tensorcnt(0);
  }

  // zero h buffer 0 region [128,256) of each row
  for (int i = tid; i < BATCH * HID; i += 512) {
    int r = i >> 7, c = i & 127;
    xh[r * XH2 + 128 + c] = (f16)0.f;
  }

  // --- async embedding prefetch helper: 32 B per thread (2 x b128) --------
  const unsigned long long embb = (unsigned long long)(uintptr_t)embh;
  auto issue_x = [&](int buf, int tok) {
    unsigned lds0 = (unsigned)(uintptr_t)(xh + row4 * XH2 + buf * 64 + seg * 16);
    unsigned long long ga = embb + (unsigned long long)tok * 128ull
                                 + (unsigned)(seg * 32);
    asm volatile("global_load_async_to_lds_b128 %0, %1, off"
                 :: "v"(lds0), "v"(ga) : "memory");
    asm volatile("global_load_async_to_lds_b128 %0, %1, off offset:16"
                 :: "v"(lds0), "v"(ga) : "memory");
  };

  // prologue: x for t=0 async, token for t=1 prefetched to register
  int tok = sent[row4 * T_LEN + (dir ? (T_LEN - 1) : 0)];
  issue_x(0, tok);
  int tokN = sent[row4 * T_LEN + (dir ? (T_LEN - 2) : 1)];
  asm volatile("s_wait_asynccnt 0x0" ::: "memory");
  __syncthreads();

  // per-wave persistent state: 4 (m,j) pairs
  float bias_r[4][4];
  float cst[4][8];
#pragma unroll
  for (int p = 0; p < 4; ++p) {
    int pr = wave * 4 + p, j = pr & 7;
#pragma unroll
    for (int g = 0; g < 4; ++g) bias_r[p][g] = bg[j * 64 + g * 16 + lq];
#pragma unroll
    for (int r = 0; r < 8; ++r) cst[p][r] = 0.f;
  }

  for (int t = 0; t < T_LEN; ++t) {
    const int tt = dir ? (T_LEN - 1 - t) : t;
    // kick off next step's embedding copy before computing this step
    if (t + 1 < T_LEN) issue_x((t + 1) & 1, tokN);
    if (t + 2 < T_LEN) {
      int ttn = dir ? (T_LEN - 3 - t) : (t + 2);
      tokN = sent[row4 * T_LEN + ttn];
    }

    const int xr = (t & 1) * 64;                // read x_t
    const int hr = 128 + (t & 1) * 128;         // read h_{t-1}
    const int hw = 128 + ((t + 1) & 1) * 128;   // write h_t

#pragma unroll
    for (int p = 0; p < 4; ++p) {
      int pr = wave * 4 + p, m = pr >> 3, j = pr & 7;
      v8f acc[4];
      const int arow = m * 16 + lq;
#pragma unroll
      for (int kc = 0; kc < 6; ++kc) {
        int kb = (kc < 2) ? (xr + kc * 32) : (hr + (kc - 2) * 32);
        // A: 16x32 f16, lane=M, two contiguous 8-f16 K segments
        const f16* ap = xh + arow * XH2 + kb + half * 8;
        v16h A;
        {
          v8h s0 = *(const v8h*)ap;
          v8h s1 = *(const v8h*)(ap + 16);
#pragma unroll
          for (int i = 0; i < 8; ++i) { A[i] = s0[i]; A[8 + i] = s1[i]; }
        }
#pragma unroll
        for (int g = 0; g < 4; ++g) {
          int zc = j * 64 + g * 16 + lq;
          // B: 32x16 f16, lane=N, two contiguous 8-f16 K segments
          const f16* bpn = Wl + zc * KDIM + kc * 32 + half * 8;
          v16h Bm;
          {
            v8h s0 = *(const v8h*)bpn;
            v8h s1 = *(const v8h*)(bpn + 16);
#pragma unroll
            for (int i = 0; i < 8; ++i) { Bm[i] = s0[i]; Bm[8 + i] = s1[i]; }
          }
          if (kc == 0) {
            v8f zc8 = {};  // inline SRC2=0 form: no accumulator init moves
            acc[g] = __builtin_amdgcn_wmma_f32_16x16x32_f16(
                false, A, false, Bm, (short)0, zc8, false, false);
          } else {
            acc[g] = __builtin_amdgcn_wmma_f32_16x16x32_f16(
                false, A, false, Bm, (short)0, acc[g], false, false);
          }
        }
      }
      // elementwise LSTM cell update; bias folded in here (replaces mov chains)
#pragma unroll
      for (int r = 0; r < 8; ++r) {
        float iv  = acc[0][r] + bias_r[p][0];
        float fvv = acc[1][r] + bias_r[p][1];
        float gv  = acc[2][r] + bias_r[p][2];
        float ov  = acc[3][r] + bias_r[p][3];
        float si = 1.f / (1.f + __expf(-iv));
        float sf = 1.f / (1.f + __expf(-fvv));
        float so = 1.f / (1.f + __expf(-ov));
        float c  = sf * cst[p][r] + si * tanhf(gv);
        cst[p][r] = c;
        float h  = so * tanhf(c);
        int M    = half * 8 + r;
        int brow = m * 16 + M;
        int unit = j * 16 + lq;
        xh[brow * XH2 + hw + unit] = (f16)h;
        hg[((size_t)tt * BATCH + brow) * HID + unit] = (f16)h;
      }
    }
    asm volatile("s_wait_asynccnt 0x0" ::: "memory");
    __syncthreads();
  }
}

// ---------------------------------------------------------------------------
// Kernel 3: feats = [h_f|h_b] @ W_out.T + b_out   (65536 x 32, K=256) WMMA.
// ---------------------------------------------------------------------------
__global__ __launch_bounds__(256) void feats_kernel(
    const f16* __restrict__ h_out, const float* __restrict__ W_out,
    const float* __restrict__ b_out, float* __restrict__ feats)
{
  const int wave = threadIdx.x >> 5, lane = threadIdx.x & 31;
  const int half = lane >> 4, lq = lane & 15;
  const int rt   = blockIdx.x * 8 + wave;
  const int arow = rt * 16 + lq;
  const int b = arow >> 9, t = arow & 511;
  const f16* hrow0 = h_out + ((size_t)t * BATCH + b) * HID;
  const f16* hrow1 = hrow0 + (size_t)T_LEN * BATCH * HID;

#pragma unroll
  for (int nt = 0; nt < 2; ++nt) {
    int n0 = nt * 16;
    float bv = b_out[n0 + lq];
    v8f acc;
#pragma unroll
    for (int kc = 0; kc < 8; ++kc) {
      const f16* src = (kc < 4) ? (hrow0 + kc * 32) : (hrow1 + (kc - 4) * 32);
      const f16* ap  = src + half * 8;
      v16h A;
      {
        v8h s0 = *(const v8h*)ap;
        v8h s1 = *(const v8h*)(ap + 16);
#pragma unroll
        for (int i = 0; i < 8; ++i) { A[i] = s0[i]; A[8 + i] = s1[i]; }
      }
      int n = n0 + lq;
      const float* wr = W_out + n * 256 + kc * 32 + half * 8;
      v16h Bm;
#pragma unroll
      for (int i = 0; i < 16; ++i) Bm[i] = (f16)wr[(i < 8) ? i : (i + 8)];
      if (kc == 0) {
        v8f zc8 = {};
        acc = __builtin_amdgcn_wmma_f32_16x16x32_f16(
            false, A, false, Bm, (short)0, zc8, false, false);
      } else {
        acc = __builtin_amdgcn_wmma_f32_16x16x32_f16(
            false, A, false, Bm, (short)0, acc, false, false);
      }
    }
#pragma unroll
    for (int r = 0; r < 8; ++r) {
      int row = rt * 16 + half * 8 + r;
      feats[(size_t)row * NTAG + n0 + lq] = acc[r] + bv;
    }
  }
}

// ---------------------------------------------------------------------------
// Kernel 4: CRF forward + gold score.  One wave32 per batch row, lane = tag.
// ---------------------------------------------------------------------------
__global__ void zero_kernel(float* out) {
  if (threadIdx.x == 0 && blockIdx.x == 0) out[0] = 0.f;
}

__global__ __launch_bounds__(256) void crf_kernel(
    const float* __restrict__ feats, const float* __restrict__ trans,
    const int* __restrict__ tags, float* __restrict__ out)
{
  const int wave = threadIdx.x >> 5, lane = threadIdx.x & 31;
  const int b = blockIdx.x * 8 + wave;

  float trow[32];
#pragma unroll
  for (int k = 0; k < 32; ++k) trow[k] = trans[lane * 32 + k];

  float fv = (lane == 0) ? 0.f : NEGV;
  for (int t = 0; t < T_LEN; ++t) {
    float e = feats[((size_t)b * T_LEN + t) * NTAG + lane];
    float m = -3.0e38f, s = 0.f;
#pragma unroll
    for (int k = 0; k < 32; ++k) {
      float v  = __shfl(fv, k, 32) + trow[k];
      float nm = fmaxf(m, v);
      s = s * __expf(m - nm) + __expf(v - nm);
      m = nm;
    }
    fv = e + m + __logf(s);
  }
  // logZ_b = logsumexp_j(fv[j] + trans[0][j])
  float v = fv + trans[lane];
  float m = v;
#pragma unroll
  for (int off = 16; off > 0; off >>= 1) m = fmaxf(m, __shfl_xor(m, off, 32));
  float s = __expf(v - m);
#pragma unroll
  for (int off = 16; off > 0; off >>= 1) s += __shfl_xor(s, off, 32);
  float logZ = m + __logf(s);

  // gold score (lanes stride over t)
  float g = 0.f;
  for (int t = lane; t < T_LEN; t += 32) {
    int tag  = tags[b * T_LEN + t];
    int prev = (t == 0) ? 0 : tags[b * T_LEN + t - 1];
    g += trans[tag * 32 + prev] + feats[((size_t)b * T_LEN + t) * NTAG + tag];
    if (t == T_LEN - 1) g += trans[tag];  // trans[0][tag_last]
  }
#pragma unroll
  for (int off = 16; off > 0; off >>= 1) g += __shfl_xor(g, off, 32);

  if (lane == 0) atomicAdd(out, logZ - g);
}

// ---------------------------------------------------------------------------
extern "C" void kernel_launch(void* const* d_in, const int* in_sizes, int n_in,
                              void* d_out, int out_size, void* d_ws, size_t ws_size,
                              hipStream_t stream) {
  (void)in_sizes; (void)n_in; (void)out_size; (void)ws_size;
  const int*   sent  = (const int*)d_in[0];
  const int*   tags  = (const int*)d_in[1];
  const float* emb   = (const float*)d_in[2];
  const float* Wih_f = (const float*)d_in[3];
  const float* Whh_f = (const float*)d_in[4];
  const float* b_f   = (const float*)d_in[5];
  const float* Wih_b = (const float*)d_in[6];
  const float* Whh_b = (const float*)d_in[7];
  const float* b_b   = (const float*)d_in[8];
  const float* W_out = (const float*)d_in[9];
  const float* b_out = (const float*)d_in[10];
  const float* trans = (const float*)d_in[11];
  float* out = (float*)d_out;

  char* ws = (char*)d_ws;
  f16*   Wp    = (f16*)ws;                              // 2*512*192 f16 = 384KB
  float* bp    = (float*)(ws + 393216);                 // 2*512 f32
  f16*   embh  = (f16*)(ws + 397312);                   // 32000*64 f16 = 4MB
  f16*   h_out = (f16*)(ws + 397312 + 4096000);         // 2*512*128*128 f16 = 32MB
  float* feats = (float*)(ws + 397312 + 4096000 + 33554432);  // 8MB

  const size_t lstm_lds = 512 * KDIM * 2 + BATCH * XH2 * 2;   // 294912 B
  (void)hipFuncSetAttribute((const void*)lstm_kernel,
                            hipFuncAttributeMaxDynamicSharedMemorySize,
                            (int)lstm_lds);

  prep_kernel<<<1024, 192, 0, stream>>>(Wih_f, Whh_f, b_f, Wih_b, Whh_b, b_b, Wp, bp);
  emb_prep_kernel<<<32000 * EMB / 256, 256, 0, stream>>>(emb, embh);
  lstm_kernel<<<2, 512, lstm_lds, stream>>>(sent, embh, Wp, bp, h_out);
  feats_kernel<<<512, 256, 0, stream>>>(h_out, W_out, b_out, feats);
  zero_kernel<<<1, 64, 0, stream>>>(out);
  crf_kernel<<<16, 256, 0, stream>>>(feats, trans, tags, out);
}
